// Agent_37984690765937
// MI455X (gfx1250) — matrix operand order repaired and checked by reading, not compiled, not run on hardware
//
#include <hip/hip_runtime.h>
#include <math.h>

#define OBSD 128
#define ACTD 18
#define HD   512
#define TD   128
#define BD   256
#define ND   (TD*BD)
#define G4   (4*HD)

typedef __attribute__((ext_vector_type(16))) _Float16 v16h;
typedef __attribute__((ext_vector_type(8)))  _Float16 v8h;
typedef __attribute__((ext_vector_type(8)))  float    v8f;

// ---- WMMA fragment loaders (CDNA5 wave32 16x16x32 f16 layouts), all-f16 operands ----
// A (16x32): lanes 0-15 row M=ln, K = 0..15; lanes 16-31 same rows, K = 16..31.
__device__ __forceinline__ v16h load_a_f16(const _Float16* __restrict__ A, int lda, int row,
                                           int k0, int half) {
  const _Float16* p = A + (size_t)row * lda + k0 + half * 16;
  v8h lo = *(const v8h*)(p);
  v8h hi = *(const v8h*)(p + 8);
  v16h r;
#pragma unroll
  for (int e = 0; e < 8; ++e) { r[e] = lo[e]; r[8 + e] = hi[e]; }
  return r;
}

// B (32x16): lanes 0-15 col N=ln, K=0..15; lanes 16-31 K=16..31.
// Bt[n*K + k] = W[k,n] -> one contiguous 32B load per lane.
__device__ __forceinline__ v16h load_b_f16(const _Float16* __restrict__ Bt, int K,
                                           int col, int k0, int half) {
  return *(const v16h*)(Bt + (size_t)col * K + k0 + half * 16);
}

#define WMMA_F16(A_, B_, C_) \
  __builtin_amdgcn_wmma_f32_16x16x32_f16(false, (A_), false, (B_), (short)0, (C_), false, false)

__device__ __forceinline__ float sigmoidf_(float x) { return 1.0f / (1.0f + __expf(-x)); }

// ---- f32 -> f16 bulk convert ----
__global__ void cvt_f16_kernel(const float* __restrict__ src, _Float16* __restrict__ dst, int n) {
  int i = blockIdx.x * blockDim.x + threadIdx.x;
  if (i < n) dst[i] = (_Float16)src[i];
}

// ---- convert + transpose weights f32[K,Nc] -> f16[Nc,K] ----
__global__ void conv_t_kernel(const float* __restrict__ W, _Float16* __restrict__ Wt,
                              int K, int Nc) {
  int i = blockIdx.x * blockDim.x + threadIdx.x;
  if (i < K * Nc) {
    int k = i / Nc, n = i % Nc;
    Wt[(size_t)n * K + k] = (_Float16)W[i];
  }
}

// ---- C16 = relu_f16(A16[M,K] @ W + bias); 2x2 16x16 tiles per wave ----
__global__ void gemm_bias_relu_f16(const _Float16* __restrict__ A, int lda,
                                   const _Float16* __restrict__ Bt, int K,
                                   const float* __restrict__ bias,
                                   _Float16* __restrict__ C, int ldc, int wnCount) {
  int lane = threadIdx.x & 31, wave = threadIdx.x >> 5;
  int half = lane >> 4, ln = lane & 15;
  int wt = blockIdx.x * (blockDim.x >> 5) + wave;
  int wm = wt / wnCount, wn = wt % wnCount;
  int mb = wm * 32, nb = wn * 32;
  v8f a00 = {}, a01 = {}, a10 = {}, a11 = {};
  for (int k0 = 0; k0 < K; k0 += 32) {
    v16h fa0 = load_a_f16(A, lda, mb + ln, k0, half);
    v16h fa1 = load_a_f16(A, lda, mb + 16 + ln, k0, half);
    v16h fb0 = load_b_f16(Bt, K, nb + ln, k0, half);
    v16h fb1 = load_b_f16(Bt, K, nb + 16 + ln, k0, half);
    a00 = WMMA_F16(fa0, fb0, a00);
    a01 = WMMA_F16(fa0, fb1, a01);
    a10 = WMMA_F16(fa1, fb0, a10);
    a11 = WMMA_F16(fa1, fb1, a11);
  }
  int nc0 = nb + ln, nc1 = nb + 16 + ln;
  float bv0 = bias[nc0], bv1 = bias[nc1];
#pragma unroll
  for (int v = 0; v < 8; ++v) {
    int r0 = mb + v + 8 * half, r1 = r0 + 16;
    C[(size_t)r0 * ldc + nc0] = (_Float16)fmaxf(a00[v] + bv0, 0.0f);
    C[(size_t)r0 * ldc + nc1] = (_Float16)fmaxf(a01[v] + bv1, 0.0f);
    C[(size_t)r1 * ldc + nc0] = (_Float16)fmaxf(a10[v] + bv0, 0.0f);
    C[(size_t)r1 * ldc + nc1] = (_Float16)fmaxf(a11[v] + bv1, 0.0f);
  }
}

// ---- per-step masked f16 copy of recurrent state: hm16 = f16(h * (1-done_t)) ----
__global__ void mask_h16_kernel(const float* __restrict__ hA, const float* __restrict__ hC,
                                const float* __restrict__ done, int t,
                                _Float16* __restrict__ hmA, _Float16* __restrict__ hmC) {
  int idx = blockIdx.x * blockDim.x + threadIdx.x;   // 0 .. 2*B*H
  int z = idx >> 17;                                  // B*H = 131072
  int r = idx & ((1 << 17) - 1);
  int b = r >> 9;
  float m = 1.0f - done[t * BD + b];
  if (z) hmC[r] = (_Float16)(hC[r] * m);
  else   hmA[r] = (_Float16)(hA[r] * m);
}

// ---- gates = x16 @ Wih + hm16 @ Whh + bih + bhh ; M=256, N=2048, K=512+512; 2x2 tiles/wave ----
__global__ void lstm_gates(const _Float16* __restrict__ X16, const _Float16* __restrict__ Hm16,
                           const _Float16* __restrict__ WihT, const _Float16* __restrict__ WhhT,
                           const float* __restrict__ bih, const float* __restrict__ bhh,
                           float* __restrict__ gates) {
  int lane = threadIdx.x & 31, wave = threadIdx.x >> 5;
  int half = lane >> 4, ln = lane & 15;
  int wt = blockIdx.x * 8 + wave;            // 8 x 64 = 512 wave-tiles, grid=64
  int wm = wt >> 6, wn = wt & 63;
  int mb = wm * 32, nb = wn * 32;
  v8f a00 = {}, a01 = {}, a10 = {}, a11 = {};
  for (int k0 = 0; k0 < HD; k0 += 32) {      // input contribution
    v16h fa0 = load_a_f16(X16, HD, mb + ln, k0, half);
    v16h fa1 = load_a_f16(X16, HD, mb + 16 + ln, k0, half);
    v16h fb0 = load_b_f16(WihT, HD, nb + ln, k0, half);
    v16h fb1 = load_b_f16(WihT, HD, nb + 16 + ln, k0, half);
    a00 = WMMA_F16(fa0, fb0, a00);
    a01 = WMMA_F16(fa0, fb1, a01);
    a10 = WMMA_F16(fa1, fb0, a10);
    a11 = WMMA_F16(fa1, fb1, a11);
  }
  for (int k0 = 0; k0 < HD; k0 += 32) {      // recurrent contribution (pre-masked f16)
    v16h fa0 = load_a_f16(Hm16, HD, mb + ln, k0, half);
    v16h fa1 = load_a_f16(Hm16, HD, mb + 16 + ln, k0, half);
    v16h fb0 = load_b_f16(WhhT, HD, nb + ln, k0, half);
    v16h fb1 = load_b_f16(WhhT, HD, nb + 16 + ln, k0, half);
    a00 = WMMA_F16(fa0, fb0, a00);
    a01 = WMMA_F16(fa0, fb1, a01);
    a10 = WMMA_F16(fa1, fb0, a10);
    a11 = WMMA_F16(fa1, fb1, a11);
  }
  int nc0 = nb + ln, nc1 = nb + 16 + ln;
  float bv0 = bih[nc0] + bhh[nc0];
  float bv1 = bih[nc1] + bhh[nc1];
#pragma unroll
  for (int v = 0; v < 8; ++v) {
    int r0 = mb + v + 8 * half, r1 = r0 + 16;
    gates[(size_t)r0 * G4 + nc0] = a00[v] + bv0;
    gates[(size_t)r0 * G4 + nc1] = a01[v] + bv1;
    gates[(size_t)r1 * G4 + nc0] = a10[v] + bv0;
    gates[(size_t)r1 * G4 + nc1] = a11[v] + bv1;
  }
}

// ---- LSTM cell elementwise for both branches (PyTorch gate order i,f,g,o) ----
__global__ void lstm_elem(const float* __restrict__ gA, const float* __restrict__ gC,
                          float* __restrict__ hA, float* __restrict__ cA,
                          float* __restrict__ hC, float* __restrict__ cC,
                          const float* __restrict__ done, int t) {
  int idx = blockIdx.x * blockDim.x + threadIdx.x;   // 0 .. 2*B*H
  int z = idx >> 17;
  int r = idx & ((1 << 17) - 1);
  int b = r >> 9, j = r & (HD - 1);
  const float* g = z ? gC : gA;
  float* hp = z ? hC : hA;
  float* cp = z ? cC : cA;
  size_t gb = (size_t)b * G4;
  float gi = g[gb + j], gf = g[gb + HD + j], gg = g[gb + 2 * HD + j], go = g[gb + 3 * HD + j];
  float m = 1.0f - done[t * BD + b];
  float c = cp[r] * m;
  c = sigmoidf_(gf) * c + sigmoidf_(gi) * tanhf(gg);
  float h = sigmoidf_(go) * tanhf(c);
  cp[r] = c;
  hp[r] = h;
}

// ---- heads: actor log-softmax/logp/entropy + critic value; one wave per batch row ----
__global__ void heads_kernel(const float* __restrict__ hA, const float* __restrict__ hC,
                             const float* __restrict__ Wact, const float* __restrict__ bact,
                             const float* __restrict__ Wcrit, const float* __restrict__ bcrit,
                             const int* __restrict__ action, float* __restrict__ out, int t) {
  int b = blockIdx.x, lane = threadIdx.x;
  int n = t * BD + b;
  const float* ha = hA + (size_t)b * HD;
  const float* hc = hC + (size_t)b * HD;
  float hr[16], hcr[16];
#pragma unroll
  for (int i = 0; i < 16; ++i) { hr[i] = ha[lane + 32 * i]; hcr[i] = hc[lane + 32 * i]; }
  float logits[ACTD];
#pragma unroll
  for (int j = 0; j < ACTD; ++j) {
    float s = 0.0f;
#pragma unroll
    for (int i = 0; i < 16; ++i) s += hr[i] * Wact[(size_t)(lane + 32 * i) * ACTD + j];
    for (int off = 16; off > 0; off >>= 1) s += __shfl_xor(s, off, 32);
    logits[j] = s + bact[j];
  }
  float mx = logits[0];
#pragma unroll
  for (int j = 1; j < ACTD; ++j) mx = fmaxf(mx, logits[j]);
  float se = 0.0f;
#pragma unroll
  for (int j = 0; j < ACTD; ++j) se += __expf(logits[j] - mx);
  float lse = mx + __logf(se);
  float ent = 0.0f;
#pragma unroll
  for (int j = 0; j < ACTD; ++j) { float lp = logits[j] - lse; ent -= __expf(lp) * lp; }
  float sv = 0.0f;
#pragma unroll
  for (int i = 0; i < 16; ++i) sv += hcr[i] * Wcrit[lane + 32 * i];
  for (int off = 16; off > 0; off >>= 1) sv += __shfl_xor(sv, off, 32);
  if (lane == 0) {
    int a = action[n];
    out[n] = logits[a] - lse;          // logp
    out[ND + n] = ent;                 // entropy
    out[2 * ND + n] = sv + bcrit[0];   // value
  }
}

__global__ void init_states(const float* __restrict__ h0a, const float* __restrict__ c0a,
                            const float* __restrict__ h0c, const float* __restrict__ c0c,
                            float* hA, float* cA, float* hC, float* cC) {
  int i = blockIdx.x * blockDim.x + threadIdx.x;
  if (i < BD * HD) { hA[i] = h0a[i]; cA[i] = c0a[i]; hC[i] = h0c[i]; cC[i] = c0c[i]; }
}

__global__ void finalize_states(const float* __restrict__ hA, const float* __restrict__ cA,
                                const float* __restrict__ hC, const float* __restrict__ cC,
                                float* __restrict__ out) {
  int i = blockIdx.x * blockDim.x + threadIdx.x;
  if (i < BD * HD) {
    float* o = out + 3 * (size_t)ND;
    o[i] = hA[i];
    o[(size_t)BD * HD + i] = cA[i];
    o[2 * (size_t)BD * HD + i] = hC[i];
    o[3 * (size_t)BD * HD + i] = cC[i];
  }
}

extern "C" void kernel_launch(void* const* d_in, const int* in_sizes, int n_in,
                              void* d_out, int out_size, void* d_ws, size_t ws_size,
                              hipStream_t stream) {
  (void)in_sizes; (void)n_in; (void)out_size; (void)ws_size;
  const float* x      = (const float*)d_in[0];
  const float* done   = (const float*)d_in[1];
  const int*   action = (const int*)  d_in[2];
  const float* h0a = (const float*)d_in[3];
  const float* c0a = (const float*)d_in[4];
  const float* h0c = (const float*)d_in[5];
  const float* c0c = (const float*)d_in[6];
  const float* W1a  = (const float*)d_in[7];  const float* b1a  = (const float*)d_in[8];
  const float* W2a  = (const float*)d_in[9];  const float* b2a  = (const float*)d_in[10];
  const float* Wih_a= (const float*)d_in[11]; const float* Whh_a= (const float*)d_in[12];
  const float* bih_a= (const float*)d_in[13]; const float* bhh_a= (const float*)d_in[14];
  const float* Wact = (const float*)d_in[15]; const float* bact = (const float*)d_in[16];
  const float* W1c  = (const float*)d_in[17]; const float* b1c  = (const float*)d_in[18];
  const float* W2c  = (const float*)d_in[19]; const float* b2c  = (const float*)d_in[20];
  const float* Wih_c= (const float*)d_in[21]; const float* Whh_c= (const float*)d_in[22];
  const float* bih_c= (const float*)d_in[23]; const float* bhh_c= (const float*)d_in[24];
  const float* Wcrit= (const float*)d_in[25]; const float* bcrit= (const float*)d_in[26];
  float* out = (float*)d_out;

  char* ws = (char*)d_ws;
  size_t off = 0;
  auto alloc = [&](size_t bytes) -> void* {
    void* p = (void*)(ws + off);
    off += (bytes + 255) & ~(size_t)255;
    return p;
  };
  _Float16* W1aT  = (_Float16*)alloc((size_t)HD * OBSD * 2);
  _Float16* W2aT  = (_Float16*)alloc((size_t)HD * HD * 2);
  _Float16* WihaT = (_Float16*)alloc((size_t)G4 * HD * 2);
  _Float16* WhhaT = (_Float16*)alloc((size_t)G4 * HD * 2);
  _Float16* W1cT  = (_Float16*)alloc((size_t)HD * OBSD * 2);
  _Float16* W2cT  = (_Float16*)alloc((size_t)HD * HD * 2);
  _Float16* WihcT = (_Float16*)alloc((size_t)G4 * HD * 2);
  _Float16* WhhcT = (_Float16*)alloc((size_t)G4 * HD * 2);
  _Float16* x16   = (_Float16*)alloc((size_t)ND * OBSD * 2);
  _Float16* h116  = (_Float16*)alloc((size_t)BD * HD * 2);
  _Float16* hma16 = (_Float16*)alloc((size_t)BD * HD * 2);
  _Float16* hmc16 = (_Float16*)alloc((size_t)BD * HD * 2);
  _Float16* hmA16 = (_Float16*)alloc((size_t)BD * HD * 2);   // masked recurrent f16, actor
  _Float16* hmC16 = (_Float16*)alloc((size_t)BD * HD * 2);   // masked recurrent f16, critic
  float* gatesA = (float*)alloc((size_t)BD * G4 * 4);
  float* gatesC = (float*)alloc((size_t)BD * G4 * 4);
  float* hA = (float*)alloc((size_t)BD * HD * 4);
  float* cA = (float*)alloc((size_t)BD * HD * 4);
  float* hC = (float*)alloc((size_t)BD * HD * 4);
  float* cC = (float*)alloc((size_t)BD * HD * 4);

  // prep: x -> f16, weights -> transposed f16 (L2-resident afterwards)
  cvt_f16_kernel<<<(ND * OBSD + 255) / 256, 256, 0, stream>>>(x, x16, ND * OBSD);
  conv_t_kernel<<<(OBSD * HD + 255) / 256, 256, 0, stream>>>(W1a, W1aT, OBSD, HD);
  conv_t_kernel<<<(HD * HD + 255) / 256, 256, 0, stream>>>(W2a, W2aT, HD, HD);
  conv_t_kernel<<<(HD * G4 + 255) / 256, 256, 0, stream>>>(Wih_a, WihaT, HD, G4);
  conv_t_kernel<<<(HD * G4 + 255) / 256, 256, 0, stream>>>(Whh_a, WhhaT, HD, G4);
  conv_t_kernel<<<(OBSD * HD + 255) / 256, 256, 0, stream>>>(W1c, W1cT, OBSD, HD);
  conv_t_kernel<<<(HD * HD + 255) / 256, 256, 0, stream>>>(W2c, W2cT, HD, HD);
  conv_t_kernel<<<(HD * G4 + 255) / 256, 256, 0, stream>>>(Wih_c, WihcT, HD, G4);
  conv_t_kernel<<<(HD * G4 + 255) / 256, 256, 0, stream>>>(Whh_c, WhhcT, HD, G4);
  init_states<<<(BD * HD + 255) / 256, 256, 0, stream>>>(h0a, c0a, h0c, c0c, hA, cA, hC, cC);

  // sequential scan over T; all launches on `stream`, deterministic order
  for (int t = 0; t < TD; ++t) {
    const _Float16* xt = x16 + (size_t)t * BD * OBSD;
    // MLPs: [256,128]@[128,512] relu -> [256,512]@[512,512] relu, f16 out
    gemm_bias_relu_f16<<<16, 256, 0, stream>>>(xt, OBSD, W1aT, OBSD, b1a, h116, HD, HD / 32);
    gemm_bias_relu_f16<<<16, 256, 0, stream>>>(h116, HD, W2aT, HD, b2a, hma16, HD, HD / 32);
    gemm_bias_relu_f16<<<16, 256, 0, stream>>>(xt, OBSD, W1cT, OBSD, b1c, h116, HD, HD / 32);
    gemm_bias_relu_f16<<<16, 256, 0, stream>>>(h116, HD, W2cT, HD, b2c, hmc16, HD, HD / 32);
    // masked f16 recurrent operand (converted once per step, not per N-tile)
    mask_h16_kernel<<<(2 * BD * HD) / 256, 256, 0, stream>>>(hA, hC, done, t, hmA16, hmC16);
    // fused gates GEMM: pure load + wmma inner loop, 2x2 tiles per wave
    lstm_gates<<<64, 256, 0, stream>>>(hma16, hmA16, WihaT, WhhaT, bih_a, bhh_a, gatesA);
    lstm_gates<<<64, 256, 0, stream>>>(hmc16, hmC16, WihcT, WhhcT, bih_c, bhh_c, gatesC);
    // cell update for both branches
    lstm_elem<<<(2 * BD * HD) / 256, 256, 0, stream>>>(gatesA, gatesC, hA, cA, hC, cC, done, t);
    // per-step heads write logp/entropy/value rows directly to d_out
    heads_kernel<<<BD, 32, 0, stream>>>(hA, hC, Wact, bact, Wcrit, bcrit, action, out, t);
  }
  finalize_states<<<(BD * HD + 255) / 256, 256, 0, stream>>>(hA, cA, hC, cC, out);
}
